// GraphSAGE_41841571397936
// MI455X (gfx1250) — matrix operand order repaired and checked by reading, not compiled, not run on hardware
//
#include <hip/hip_runtime.h>
#include <hip/hip_bf16.h>

// GraphSAGE (2x SAGEConv, mean aggregation) for MI455X / gfx1250.
// Scatter phase: wave-per-edge fp32 atomics (accumulators are L2-resident:
//   x 12.8MB, h/agg <=25.6MB each, all << 192MB L2).
// Dense phase:   bf16 WMMA 16x16x32, f32 accumulate; 2 row-tiles per wave.

typedef __attribute__((ext_vector_type(16))) __bf16 v16bf;
typedef __attribute__((ext_vector_type(8)))  float  v8f;

#define IN_C  32
#define HID_C 64
#define OUT_C 32

__device__ __forceinline__ __bf16 f2bf(float f) {
    return static_cast<__bf16>(f);   // native v_cvt on gfx1250 (RNE)
}

// ---- WMMA operand builders (wave32, 16x16x32 bf16) ----------------------
// A (16x32, MxK): lane<16 -> row M=lane, halves 0..7 = K 0..7, 8..15 = K 16..23
//                 lane>=16 -> row M=lane-16, halves 0..7 = K 8..15, 8..15 = K 24..31
__device__ __forceinline__ v16bf build_A(const float* __restrict__ row,
                                         float scale, int lane) {
    const int koff = (lane >> 4) * 8;   // 0 or 8
    v16bf a;
#pragma unroll
    for (int j = 0; j < 8; ++j) a[j]     = f2bf(row[koff + j]      * scale);
#pragma unroll
    for (int j = 0; j < 8; ++j) a[8 + j] = f2bf(row[16 + koff + j] * scale);
    return a;
}

// B (32x16, KxN): lane<16 -> col N=lane, halves 0..7 = K 0..7, 8..15 = K 16..23
//                 lane>=16 -> col N=lane-16, halves 0..7 = K 8..15, 8..15 = K 24..31
__device__ __forceinline__ v16bf build_B(const float* __restrict__ W, int ldn,
                                         int n0, int lane) {
    const int n     = n0 + (lane & 15);
    const int kbase = (lane >> 4) * 8;  // 0 or 8
    v16bf b;
#pragma unroll
    for (int j = 0; j < 16; ++j) b[j] = f2bf(W[(size_t)(kbase + (j & 7) + (j >> 3) * 16) * ldn + n]);
    return b;
}

__device__ __forceinline__ float inv_deg(const float* __restrict__ cnt, int row) {
    return __builtin_amdgcn_rcpf(fmaxf(cnt[row], 1.0f));
}

// store one 16x16 f32 D tile (rows node0..node0+15, cols c0..c0+15) with
// bias (+ optional relu); branch-free when the tile is fully in range.
__device__ __forceinline__ void store_D(float* __restrict__ dst, int ldc,
                                        int node0, int c0, const v8f& acc,
                                        float bias, bool relu, int n_nodes,
                                        int lane) {
    const int n    = lane & 15;
    const int msel = (lane >> 4) * 8;
    float* p = dst + (size_t)(node0 + msel) * ldc + c0 + n;
    if (node0 + 16 <= n_nodes) {
#pragma unroll
        for (int r = 0; r < 8; ++r) {
            float v = acc[r] + bias;
            if (relu) v = fmaxf(v, 0.0f);
            p[(size_t)r * ldc] = v;
        }
    } else {
#pragma unroll
        for (int r = 0; r < 8; ++r) {
            if (node0 + msel + r < n_nodes) {
                float v = acc[r] + bias;
                if (relu) v = fmaxf(v, 0.0f);
                p[(size_t)r * ldc] = v;
            }
        }
    }
}

// ---- workspace zeroing (b128 stores) ------------------------------------
__global__ void zero_f32_kernel(float* __restrict__ p, long n) {
    const long n4     = n >> 2;
    long i            = (long)blockIdx.x * blockDim.x + threadIdx.x;
    const long stride = (long)gridDim.x * blockDim.x;
    float4* p4 = (float4*)p;
    for (long j = i; j < n4; j += stride) p4[j] = make_float4(0.f, 0.f, 0.f, 0.f);
    for (long j = (n4 << 2) + i; j < n; j += stride) p[j] = 0.0f;
}

// ---- layer-1 edge scatter: agg1[dst] += x[src]; cnt[dst] += 1 -----------
__global__ void scatter1_kernel(const float* __restrict__ x,
                                const long long* __restrict__ src,
                                const long long* __restrict__ dst,
                                float* __restrict__ agg1,
                                float* __restrict__ cnt, int n_edges) {
    const int wave = (blockIdx.x * blockDim.x + threadIdx.x) >> 5;
    const int lane = threadIdx.x & 31;
    if (wave >= n_edges) return;
    const int s = (int)src[wave];
    const int d = (int)dst[wave];
    // lane == channel: coalesced 128B gather + coalesced 128B atomic burst
    atomicAdd(&agg1[(size_t)d * IN_C + lane], x[(size_t)s * IN_C + lane]);
    if (lane == 0) atomicAdd(&cnt[d], 1.0f);
}

// ---- layer-2 edge scatter: agg2[dst] += h[src] (64 channels) ------------
__global__ void scatter2_kernel(const float* __restrict__ h,
                                const long long* __restrict__ src,
                                const long long* __restrict__ dst,
                                float* __restrict__ agg2, int n_edges) {
    const int wave = (blockIdx.x * blockDim.x + threadIdx.x) >> 5;
    const int lane = threadIdx.x & 31;
    if (wave >= n_edges) return;
    const int s = (int)src[wave];
    const int d = (int)dst[wave];
#pragma unroll
    for (int k = 0; k < 2; ++k) {
        const int c = lane + 32 * k;
        atomicAdd(&agg2[(size_t)d * HID_C + c], h[(size_t)s * HID_C + c]);
    }
}

// ---- layer-1 node GEMM: h = relu((agg1/deg)@W1_l + b1 + x@W1_r) ---------
// One wave handles TWO 16-node row tiles (B operands reused across both).
__global__ void node1_kernel(const float* __restrict__ x,
                             const float* __restrict__ agg1,
                             const float* __restrict__ cnt,
                             const float* __restrict__ W1l,
                             const float* __restrict__ b1,
                             const float* __restrict__ W1r,
                             float* __restrict__ h, int n_nodes, int n_tiles) {
    const int wave = (blockIdx.x * blockDim.x + threadIdx.x) >> 5;
    const int lane = threadIdx.x & 31;
    const int tile0 = wave * 2;
    if (tile0 >= n_tiles) return;
    const bool has1 = (tile0 + 1) < n_tiles;
    const int nodeA = tile0 * 16;
    const int nodeB = nodeA + 16;

    int rowA = nodeA + (lane & 15);
    if (rowA >= n_nodes) rowA = n_nodes - 1;
    int rowB = has1 ? nodeB + (lane & 15) : rowA;
    if (rowB >= n_nodes) rowB = n_nodes - 1;

    const v16bf aggA = build_A(agg1 + (size_t)rowA * IN_C, inv_deg(cnt, rowA), lane);
    const v16bf xA   = build_A(x    + (size_t)rowA * IN_C, 1.0f, lane);
    const v16bf aggB = build_A(agg1 + (size_t)rowB * IN_C, inv_deg(cnt, rowB), lane);
    const v16bf xB   = build_A(x    + (size_t)rowB * IN_C, 1.0f, lane);

#pragma unroll
    for (int t = 0; t < HID_C / 16; ++t) {          // 4 col tiles of 16
        const v16bf bl = build_B(W1l, HID_C, t * 16, lane);
        const v16bf br = build_B(W1r, HID_C, t * 16, lane);
        const float bias = b1[t * 16 + (lane & 15)];

        v8f acc = {};
        acc = __builtin_amdgcn_wmma_f32_16x16x32_bf16(false, aggA, false, bl,
                                                      (short)0, acc, false, false);
        acc = __builtin_amdgcn_wmma_f32_16x16x32_bf16(false, xA,   false, br,
                                                      (short)0, acc, false, false);
        store_D(h, HID_C, nodeA, t * 16, acc, bias, true, n_nodes, lane);

        if (has1) {
            v8f acc2 = {};
            acc2 = __builtin_amdgcn_wmma_f32_16x16x32_bf16(false, aggB, false, bl,
                                                           (short)0, acc2, false, false);
            acc2 = __builtin_amdgcn_wmma_f32_16x16x32_bf16(false, xB,   false, br,
                                                           (short)0, acc2, false, false);
            store_D(h, HID_C, nodeB, t * 16, acc2, bias, true, n_nodes, lane);
        }
    }
}

// ---- layer-2 node GEMM: out = (agg2/deg)@W2_l + b2 + h@W2_r -------------
__global__ void node2_kernel(const float* __restrict__ h,
                             const float* __restrict__ agg2,
                             const float* __restrict__ cnt,
                             const float* __restrict__ W2l,
                             const float* __restrict__ b2,
                             const float* __restrict__ W2r,
                             float* __restrict__ out, int n_nodes, int n_tiles) {
    const int wave = (blockIdx.x * blockDim.x + threadIdx.x) >> 5;
    const int lane = threadIdx.x & 31;
    const int tile0 = wave * 2;
    if (tile0 >= n_tiles) return;
    const bool has1 = (tile0 + 1) < n_tiles;
    const int nodeA = tile0 * 16;
    const int nodeB = nodeA + 16;

    int rowA = nodeA + (lane & 15);
    if (rowA >= n_nodes) rowA = n_nodes - 1;
    int rowB = has1 ? nodeB + (lane & 15) : rowA;
    if (rowB >= n_nodes) rowB = n_nodes - 1;

    const float invA = inv_deg(cnt, rowA);
    const float invB = inv_deg(cnt, rowB);
    const float* arA = agg2 + (size_t)rowA * HID_C;
    const float* hrA = h    + (size_t)rowA * HID_C;
    const float* arB = agg2 + (size_t)rowB * HID_C;
    const float* hrB = h    + (size_t)rowB * HID_C;

    // K = 64 -> two K=32 chunks per operand
    const v16bf aggA0 = build_A(arA,      invA, lane);
    const v16bf aggA1 = build_A(arA + 32, invA, lane);
    const v16bf hA0   = build_A(hrA,      1.0f, lane);
    const v16bf hA1   = build_A(hrA + 32, 1.0f, lane);
    const v16bf aggB0 = build_A(arB,      invB, lane);
    const v16bf aggB1 = build_A(arB + 32, invB, lane);
    const v16bf hB0   = build_A(hrB,      1.0f, lane);
    const v16bf hB1   = build_A(hrB + 32, 1.0f, lane);

#pragma unroll
    for (int t = 0; t < OUT_C / 16; ++t) {          // 2 col tiles of 16
        const v16bf bl0 = build_B(W2l,              OUT_C, t * 16, lane);
        const v16bf bl1 = build_B(W2l + 32 * OUT_C, OUT_C, t * 16, lane);
        const v16bf br0 = build_B(W2r,              OUT_C, t * 16, lane);
        const v16bf br1 = build_B(W2r + 32 * OUT_C, OUT_C, t * 16, lane);
        const float bias = b2[t * 16 + (lane & 15)];

        v8f acc = {};
        acc = __builtin_amdgcn_wmma_f32_16x16x32_bf16(false, aggA0, false, bl0,
                                                      (short)0, acc, false, false);
        acc = __builtin_amdgcn_wmma_f32_16x16x32_bf16(false, aggA1, false, bl1,
                                                      (short)0, acc, false, false);
        acc = __builtin_amdgcn_wmma_f32_16x16x32_bf16(false, hA0,   false, br0,
                                                      (short)0, acc, false, false);
        acc = __builtin_amdgcn_wmma_f32_16x16x32_bf16(false, hA1,   false, br1,
                                                      (short)0, acc, false, false);
        store_D(out, OUT_C, nodeA, t * 16, acc, bias, false, n_nodes, lane);

        if (has1) {
            v8f acc2 = {};
            acc2 = __builtin_amdgcn_wmma_f32_16x16x32_bf16(false, aggB0, false, bl0,
                                                           (short)0, acc2, false, false);
            acc2 = __builtin_amdgcn_wmma_f32_16x16x32_bf16(false, aggB1, false, bl1,
                                                           (short)0, acc2, false, false);
            acc2 = __builtin_amdgcn_wmma_f32_16x16x32_bf16(false, hB0,   false, br0,
                                                           (short)0, acc2, false, false);
            acc2 = __builtin_amdgcn_wmma_f32_16x16x32_bf16(false, hB1,   false, br1,
                                                           (short)0, acc2, false, false);
            store_D(out, OUT_C, nodeB, t * 16, acc2, bias, false, n_nodes, lane);
        }
    }
}

extern "C" void kernel_launch(void* const* d_in, const int* in_sizes, int n_in,
                              void* d_out, int out_size, void* d_ws, size_t ws_size,
                              hipStream_t stream) {
    const float*     x   = (const float*)d_in[0];
    const long long* ei  = (const long long*)d_in[1];   // int64 [2, E]
    const float*     W1l = (const float*)d_in[2];
    const float*     b1  = (const float*)d_in[3];
    const float*     W1r = (const float*)d_in[4];
    const float*     W2l = (const float*)d_in[5];
    const float*     b2  = (const float*)d_in[6];
    const float*     W2r = (const float*)d_in[7];
    float*           out = (float*)d_out;

    const int N = in_sizes[0] / IN_C;   // 100000
    const int E = in_sizes[1] / 2;      // 1600000
    const long long* src = ei;
    const long long* dst = ei + E;

    // Workspace layout (fp32): [cnt N][agg1 N*32][agg2 N*64][h N*64] ~64.4MB
    float* cnt  = (float*)d_ws;
    float* agg1 = cnt  + N;
    float* agg2 = agg1 + (size_t)N * IN_C;
    float* h    = agg2 + (size_t)N * HID_C;

    // 1) zero cnt + agg1 + agg2 (contiguous)
    const long zn = (long)N * (1 + IN_C + HID_C);
    zero_f32_kernel<<<1024, 256, 0, stream>>>((float*)d_ws, zn);

    // 2) layer-1 scatter: one wave per edge
    const int sblocks = (E + 7) / 8;    // 8 waves per 256-thread block
    scatter1_kernel<<<sblocks, 256, 0, stream>>>(x, src, dst, agg1, cnt, E);

    // 3) layer-1 dense (WMMA): one wave per two 16-node tiles
    const int tiles   = (N + 15) / 16;
    const int waves   = (tiles + 1) / 2;
    const int nblocks = (waves + 7) / 8;
    node1_kernel<<<nblocks, 256, 0, stream>>>(x, agg1, cnt, W1l, b1, W1r, h, N, tiles);

    // 4) layer-2 scatter
    scatter2_kernel<<<sblocks, 256, 0, stream>>>(h, src, dst, agg2, E);

    // 5) layer-2 dense (WMMA) -> out
    node2_kernel<<<nblocks, 256, 0, stream>>>(h, agg2, cnt, W2l, b2, W2r, out, N, tiles);
}
